// LSTMWithGateAccess_47012712022120
// MI455X (gfx1250) — matrix operand order repaired and checked by reading, not compiled, not run on hardware
//
#include <hip/hip_runtime.h>

// Problem sizes (fixed by the reference)
#define TT  512
#define BB  16
#define NIN 128
#define PP  256
#define HH  512
#define G4  2048                 // 4*H
#define OC  ((size_t)BB * TT * HH)
#define LDSW 520                 // padded LDS row stride (elements) to spread banks
#define LDS_ELEMS (3 * 64 * LDSW)
#define LDS_BYTES (LDS_ELEMS * 2)

typedef __attribute__((ext_vector_type(16))) __bf16 v16bf;
typedef __attribute__((ext_vector_type(8)))  float  v8f;
typedef __attribute__((ext_vector_type(4)))  unsigned v4u;

// ---------------- helpers ----------------

__device__ __forceinline__ unsigned short f2bf(float f) {
  unsigned u = __float_as_uint(f);
  unsigned rnd = 0x7FFFu + ((u >> 16) & 1u);   // round-to-nearest-even
  return (unsigned short)((u + rnd) >> 16);
}

// A-fragment (16x32 bf16, MxK): lane<16 holds K = kb+0..15 of row M=lane (contiguous);
// lane>=16 holds K = kb+16..31. Caller passes p = row*stride + kb + (hi?16:0).
__device__ __forceinline__ v16bf ld_a(const unsigned short* p) {
  union { v4u q[2]; v16bf v; } u;
  u.q[0] = *(const v4u*)(p);
  u.q[1] = *(const v4u*)(p + 8);
  return u.v;
}

// B-fragment (32x16 bf16, KxN): lane<16 holds K = kb+0..15 of column N=lane (contiguous);
// lane>=16 holds K = kb+16..31. Caller passes p = col_row*stride + kb + (hi?16:0).
// Works for both global and LDS (__shared__) base pointers.
__device__ __forceinline__ v16bf ld_b(const unsigned short* p) {
  union { v4u q[2]; v16bf v; } u;
  u.q[0] = *(const v4u*)(p);
  u.q[1] = *(const v4u*)(p + 8);
  return u.v;
}

__device__ __forceinline__ v8f bf16_mma(v16bf a, v16bf b, v8f c) {
  return __builtin_amdgcn_wmma_f32_16x16x32_bf16(
      /*neg_a=*/false, a, /*neg_b=*/false, b,
      /*c_mod=*/(short)0, c, /*reuse_a=*/false, /*reuse_b=*/false);
}

__device__ __forceinline__ float sigm(float x) { return 1.0f / (1.0f + __expf(-x)); }

// Monotonic grid barrier: each block adds 1; all blocks spin until count reaches nwg*epoch.
__device__ __forceinline__ void gridbar(unsigned* bar, unsigned target) {
  __threadfence();
  __syncthreads();
  if (threadIdx.x == 0) {
    __hip_atomic_fetch_add(bar, 1u, __ATOMIC_RELEASE, __HIP_MEMORY_SCOPE_AGENT);
    while (__hip_atomic_load(bar, __ATOMIC_ACQUIRE, __HIP_MEMORY_SCOPE_AGENT) < target) {
      __builtin_amdgcn_s_sleep(1);
    }
  }
  __syncthreads();
  __threadfence();
}

// ---------------- phase A kernels (parallel over B*T) ----------------

__global__ void cvt_kernel(const float* __restrict__ s, unsigned short* __restrict__ d, int n) {
  int i = blockIdx.x * blockDim.x + threadIdx.x;
  if (i < n) d[i] = f2bf(s[i]);
}

__global__ void addv_kernel(const float* __restrict__ a, const float* __restrict__ b,
                            float* __restrict__ o, int n) {
  int i = blockIdx.x * blockDim.x + threadIdx.x;
  if (i < n) o[i] = a[i] + b[i];
}

__global__ void zero_kernel(unsigned* __restrict__ p, int n) {
  int i = blockIdx.x * blockDim.x + threadIdx.x;
  if (i < n) p[i] = 0u;
}

// proj = x @ w_proj.T + b_proj ; M=8192(K=128) tiles 512x16, one wave per 16x16 tile
__global__ void __launch_bounds__(128)
proj_gemm_kernel(const unsigned short* __restrict__ xb,
                 const unsigned short* __restrict__ wpb,
                 const float* __restrict__ bproj,
                 unsigned short* __restrict__ projb) {
  const int lane = threadIdx.x & 31;
  const int wid  = blockIdx.x * (blockDim.x >> 5) + (threadIdx.x >> 5);
  const int r  = lane & 15;
  const int hi = lane >> 4;
  const int nt = wid & 15;
  const int mt = wid >> 4;
  const int m0 = mt * 16, n0 = nt * 16;

  v8f acc = {};
  const float bv = bproj[n0 + r];
#pragma unroll
  for (int v = 0; v < 8; ++v) acc[v] = bv;

#pragma unroll
  for (int kb = 0; kb < NIN; kb += 32) {
    v16bf a = ld_a(xb  + (size_t)(m0 + r) * NIN + kb + hi * 16);
    v16bf b = ld_b(wpb + (size_t)(n0 + r) * NIN + kb + hi * 16);
    acc = bf16_mma(a, b, acc);
  }
#pragma unroll
  for (int v = 0; v < 8; ++v)
    projb[(size_t)(m0 + hi * 8 + v) * PP + n0 + r] = f2bf(acc[v]);
}

// gates_in0[t][b][:] = proj @ w_ih0.T + (b_ih0 + b_hh0) ; M=8192(K=256), tiles 512x128
__global__ void __launch_bounds__(128)
gates0_gemm_kernel(const unsigned short* __restrict__ projb,
                   const unsigned short* __restrict__ wih0b,
                   const float* __restrict__ bias01,
                   float* __restrict__ g0) {
  const int lane = threadIdx.x & 31;
  const int wid  = blockIdx.x * (blockDim.x >> 5) + (threadIdx.x >> 5);
  const int r  = lane & 15;
  const int hi = lane >> 4;
  const int nt = wid & 127;
  const int mt = wid >> 7;
  const int m0 = mt * 16, n0 = nt * 16;

  v8f acc = {};
  const float bv = bias01[n0 + r];
#pragma unroll
  for (int v = 0; v < 8; ++v) acc[v] = bv;

#pragma unroll
  for (int kb = 0; kb < PP; kb += 32) {
    v16bf a = ld_a(projb + (size_t)(m0 + r) * PP + kb + hi * 16);
    v16bf b = ld_b(wih0b + (size_t)(n0 + r) * PP + kb + hi * 16);
    acc = bf16_mma(a, b, acc);
  }
#pragma unroll
  for (int v = 0; v < 8; ++v) {
    int m = m0 + hi * 8 + v;           // m = b*T + t  (x is (B,T,*) row-major)
    int bidx = m >> 9;                  // /512
    int tidx = m & 511;
    g0[(size_t)tidx * (BB * G4) + (size_t)bidx * G4 + n0 + r] = acc[v];
  }
}

// ---------------- phase B: persistent sequential LSTM ----------------
// 32 blocks x 32 threads = 32 waves; wave w owns hidden columns [16w, 16w+16).
// Weight rows for those columns (whh0/wih1/whh1: 3 x 64 rows x 512 bf16) are staged
// into LDS once and re-read every timestep (512x reuse at LDS latency).
// c0/c1 live in registers. h0/h1 shared via ping-pong bf16 buffers in global (L2).
// One grid barrier per phase: phase t computes layer1(t) and layer0(t+1).

__global__ void __launch_bounds__(32)
lstm_seq_kernel(const float* __restrict__ g0,
                const unsigned short* __restrict__ whh0,
                const unsigned short* __restrict__ wih1,
                const unsigned short* __restrict__ whh1,
                const float* __restrict__ bias1,
                unsigned short* __restrict__ h0buf,
                unsigned short* __restrict__ h1buf,
                float* __restrict__ out,
                unsigned* __restrict__ bar,
                unsigned nwg)
{
  extern __shared__ unsigned short lds[];
  unsigned short* lwhh0 = lds;                     // [64][LDSW]
  unsigned short* lwih1 = lds + 64 * LDSW;
  unsigned short* lwhh1 = lds + 2 * 64 * LDSW;

  const int lane = threadIdx.x & 31;
  const int wave = blockIdx.x;                     // 1 wave per block
  const int r    = lane & 15;
  const int hi   = lane >> 4;
  const int j0   = wave << 4;                      // 32 waves * 16 = H columns
  const int mrow = hi * 8;                         // C/D row base for this lane half

  // ---- stage this block's weight rows into LDS (row = s*16 + rr  <-  s*HH + j0 + rr) ----
  for (int idx = lane; idx < 64 * (HH / 8); idx += 32) {
    int row = idx >> 6;                            // 0..63  (HH/8 == 64 chunks per row)
    int kc  = (idx & 63) * 8;                      // element offset within row
    int s = row >> 4, rr = row & 15;
    size_t gofs = (size_t)(s * HH + j0 + rr) * HH + kc;
    *(v4u*)(lwhh0 + (size_t)row * LDSW + kc) = *(const v4u*)(whh0 + gofs);
    *(v4u*)(lwih1 + (size_t)row * LDSW + kc) = *(const v4u*)(wih1 + gofs);
    *(v4u*)(lwhh1 + (size_t)row * LDSW + kc) = *(const v4u*)(whh1 + gofs);
  }
  __syncthreads();

  v8f c0 = {}; v8f c1 = {};
  float b1v[4];
#pragma unroll
  for (int s = 0; s < 4; ++s) b1v[s] = bias1[s * HH + j0 + r];

  // layer0(t): h0(t-1)=buf[t&1] -> h0(t)=buf[(t+1)&1], consumes gates_in0[t]
  auto layer0 = [&](int t) {
    const unsigned short* h0r = h0buf + (size_t)(t & 1) * (BB * HH);
    unsigned short*       h0w = h0buf + (size_t)((t + 1) & 1) * (BB * HH);
    v8f acc[4];
    const float* gt = g0 + (size_t)t * (BB * G4);
#pragma unroll
    for (int s = 0; s < 4; ++s)
#pragma unroll
      for (int v = 0; v < 8; ++v)
        acc[s][v] = gt[(size_t)(mrow + v) * G4 + s * HH + j0 + r];
    for (int kb = 0; kb < HH; kb += 32) {
      v16bf a = ld_a(h0r + (size_t)r * HH + kb + hi * 16);
#pragma unroll
      for (int s = 0; s < 4; ++s) {
        v16bf b = ld_b(lwhh0 + (size_t)(s * 16 + r) * LDSW + kb + hi * 16);
        acc[s] = bf16_mma(a, b, acc[s]);
      }
    }
#pragma unroll
    for (int v = 0; v < 8; ++v) {
      float iv = sigm(acc[0][v]);
      float fv = sigm(acc[1][v]);
      float gv = tanhf(acc[2][v]);
      float ov = sigm(acc[3][v]);
      float cn = fv * c0[v] + iv * gv;
      c0[v] = cn;
      float hn = ov * tanhf(cn);
      h0w[(size_t)(mrow + v) * HH + j0 + r] = f2bf(hn);
    }
  };

  // layer1(t): reads h0(t)=buf[(t+1)&1], h1(t-1)=buf1[t&1] -> h1(t)=buf1[(t+1)&1] + outputs
  auto layer1 = [&](int t) {
    const unsigned short* h0n = h0buf + (size_t)((t + 1) & 1) * (BB * HH);
    const unsigned short* h1r = h1buf + (size_t)(t & 1) * (BB * HH);
    unsigned short*       h1w = h1buf + (size_t)((t + 1) & 1) * (BB * HH);
    v8f acc[4];
#pragma unroll
    for (int s = 0; s < 4; ++s)
#pragma unroll
      for (int v = 0; v < 8; ++v)
        acc[s][v] = b1v[s];
    for (int kb = 0; kb < HH; kb += 32) {
      v16bf a = ld_a(h0n + (size_t)r * HH + kb + hi * 16);
#pragma unroll
      for (int s = 0; s < 4; ++s) {
        v16bf b = ld_b(lwih1 + (size_t)(s * 16 + r) * LDSW + kb + hi * 16);
        acc[s] = bf16_mma(a, b, acc[s]);
      }
    }
    for (int kb = 0; kb < HH; kb += 32) {
      v16bf a = ld_a(h1r + (size_t)r * HH + kb + hi * 16);
#pragma unroll
      for (int s = 0; s < 4; ++s) {
        v16bf b = ld_b(lwhh1 + (size_t)(s * 16 + r) * LDSW + kb + hi * 16);
        acc[s] = bf16_mma(a, b, acc[s]);
      }
    }
#pragma unroll
    for (int v = 0; v < 8; ++v) {
      float iv = sigm(acc[0][v]);
      float fv = sigm(acc[1][v]);
      float gv = tanhf(acc[2][v]);
      float ov = sigm(acc[3][v]);
      float cn = fv * c1[v] + iv * gv;
      c1[v] = cn;
      float hn = ov * tanhf(cn);
      size_t o = (size_t)(mrow + v) * (TT * HH) + (size_t)t * HH + j0 + r;
      out[0 * OC + o] = hn;   // h1
      out[1 * OC + o] = cn;   // c1
      out[2 * OC + o] = iv;   // i
      out[3 * OC + o] = fv;   // f
      out[4 * OC + o] = gv;   // g
      out[5 * OC + o] = ov;   // o
      h1w[(size_t)(mrow + v) * HH + j0 + r] = f2bf(hn);
    }
  };

  unsigned epoch = 0;
  layer0(0);                               // prologue: h0(0)
  gridbar(bar, nwg * (++epoch));
  for (int t = 0; t < TT; ++t) {
    layer1(t);                             // h1(t) + outputs(t)
    if (t + 1 < TT) layer0(t + 1);         // h0(t+1)  (independent of layer1(t))
    gridbar(bar, nwg * (++epoch));         // one barrier per phase (513 total)
  }
}

// ---------------- host side ----------------

extern "C" void kernel_launch(void* const* d_in, const int* in_sizes, int n_in,
                              void* d_out, int out_size, void* d_ws, size_t ws_size,
                              hipStream_t stream) {
  (void)in_sizes; (void)n_in; (void)out_size; (void)ws_size;
  const float* x     = (const float*)d_in[0];
  const float* wproj = (const float*)d_in[1];
  const float* bproj = (const float*)d_in[2];
  const float* wih0  = (const float*)d_in[3];
  const float* whh0  = (const float*)d_in[4];
  const float* bih0  = (const float*)d_in[5];
  const float* bhh0  = (const float*)d_in[6];
  const float* wih1  = (const float*)d_in[7];
  const float* whh1  = (const float*)d_in[8];
  const float* bih1  = (const float*)d_in[9];
  const float* bhh1  = (const float*)d_in[10];

  char* ws = (char*)d_ws;
  size_t off = 0;
  auto carve = [&](size_t bytes) -> char* {
    char* p = ws + off;
    off = (off + bytes + 255) & ~((size_t)255);
    return p;
  };
  unsigned short* xb    = (unsigned short*)carve((size_t)BB * TT * NIN * 2);
  unsigned short* wpb   = (unsigned short*)carve((size_t)PP * NIN * 2);
  unsigned short* wih0b = (unsigned short*)carve((size_t)G4 * PP * 2);
  unsigned short* whh0b = (unsigned short*)carve((size_t)G4 * HH * 2);
  unsigned short* wih1b = (unsigned short*)carve((size_t)G4 * HH * 2);
  unsigned short* whh1b = (unsigned short*)carve((size_t)G4 * HH * 2);
  unsigned short* projb = (unsigned short*)carve((size_t)BB * TT * PP * 2);
  float*          g0    = (float*)carve((size_t)TT * BB * G4 * 4);   // [T][B][4H]
  float*          bias01= (float*)carve((size_t)G4 * 4);
  float*          bias1 = (float*)carve((size_t)G4 * 4);
  unsigned short* h0buf = (unsigned short*)carve((size_t)2 * BB * HH * 2);
  unsigned short* h1buf = (unsigned short*)carve((size_t)2 * BB * HH * 2);
  unsigned*       bar   = (unsigned*)carve(256);

  // bf16 conversions of x and all weight matrices
  auto cvt = [&](const float* s, unsigned short* d, int n) {
    cvt_kernel<<<(n + 255) / 256, 256, 0, stream>>>(s, d, n);
  };
  cvt(x,     xb,    BB * TT * NIN);
  cvt(wproj, wpb,   PP * NIN);
  cvt(wih0,  wih0b, G4 * PP);
  cvt(whh0,  whh0b, G4 * HH);
  cvt(wih1,  wih1b, G4 * HH);
  cvt(whh1,  whh1b, G4 * HH);

  addv_kernel<<<(G4 + 255) / 256, 256, 0, stream>>>(bih0, bhh0, bias01, G4);
  addv_kernel<<<(G4 + 255) / 256, 256, 0, stream>>>(bih1, bhh1, bias1,  G4);

  // zero initial hidden states and the barrier counter (kernel-based: graph-capture safe)
  zero_kernel<<<((2 * BB * HH / 2) + 255) / 256, 256, 0, stream>>>((unsigned*)h0buf, 2 * BB * HH / 2);
  zero_kernel<<<((2 * BB * HH / 2) + 255) / 256, 256, 0, stream>>>((unsigned*)h1buf, 2 * BB * HH / 2);
  zero_kernel<<<1, 64, 0, stream>>>(bar, 64);

  // phase A GEMMs (WMMA)
  proj_gemm_kernel  <<<(BB * TT / 16) * (PP / 16) / 4, 128, 0, stream>>>(xb, wpb, bproj, projb);
  gates0_gemm_kernel<<<(BB * TT / 16) * (G4 / 16) / 4, 128, 0, stream>>>(projb, wih0b, bias01, g0);

  // phase B: persistent recurrent kernel, 1 wave/WGP, weights LDS-resident
  lstm_seq_kernel<<<32, 32, LDS_BYTES, stream>>>(g0, whh0b, wih1b, whh1b, bias1,
                                                 h0buf, h1buf, (float*)d_out, bar, 32u);
}
